// VisionLlamaPyramidBlock_44607530336835
// MI455X (gfx1250) — compile-verified, matrix operand order checked by reading, not run on hardware
//
#include <hip/hip_runtime.h>
#include <math.h>

typedef __attribute__((ext_vector_type(16))) __bf16   v16bf;
typedef __attribute__((ext_vector_type(8)))  float    v8f;
typedef __attribute__((ext_vector_type(4)))  unsigned v4u;

#define DIMD   768
#define HEADS  12
#define HD     64
#define MLP    3072
#define NTOK   1024
#define BATCH  32
#define MROWS  (BATCH * NTOK)      // 32768
#define QKVN   (DIMD + 2 * HD)     // 896

// ---------- helpers ----------
static __device__ __forceinline__ __bf16 to_bf16(float f) {
  unsigned u = __builtin_bit_cast(unsigned, f);
  u += 0x7FFFu + ((u >> 16) & 1u);             // round-to-nearest-even
  unsigned short h = (unsigned short)(u >> 16);
  return __builtin_bit_cast(__bf16, h);
}

union Frag { v16bf v; v4u u[2]; };

// Load one 16x32 bf16 WMMA fragment (A row-major tile, or column tile of B via B^T).
// ISA layout: lanes 0-15 -> rows 0-15, K = 0..7 & 16..23 ; lanes 16-31 -> K = 8..15 & 24..31.
static __device__ __forceinline__ v16bf load_frag(const __bf16* __restrict__ base,
                                                  int ld, int row, int k, int lane) {
  const __bf16* p = base + (size_t)(row + (lane & 15)) * ld + (k + ((lane >> 4) << 3));
  Frag f;
  f.u[0] = *reinterpret_cast<const v4u*>(p);
  f.u[1] = *reinterpret_cast<const v4u*>(p + 16);
  return f.v;
}

// ---------- weight pack: W (K x N, f32) -> W^T (N x K, bf16) via LDS tile ----------
__global__ __launch_bounds__(256) void k_transpose_pack(const float* __restrict__ W,
                                                        __bf16* __restrict__ Wt,
                                                        int K, int N) {
  __shared__ float tile[32][33];
  int bn = blockIdx.x * 32, bk = blockIdx.y * 32;
  int tx = threadIdx.x, ty = threadIdx.y;            // block (32, 8)
  for (int r = ty; r < 32; r += 8)
    tile[r][tx] = W[(size_t)(bk + r) * N + (bn + tx)];
  __syncthreads();
  for (int r = ty; r < 32; r += 8)
    Wt[(size_t)(bn + r) * K + (bk + tx)] = to_bf16(tile[tx][r]);
}

// ---------- axial rotary sin table (1024 x 768), bf16 ----------
__global__ __launch_bounds__(256) void k_sin(__bf16* __restrict__ S) {
  int t = blockIdx.x;                 // 0..1023
  int i = t >> 5, j = t & 31;
  for (int c = threadIdx.x; c < DIMD; c += 256) {
    int kk = c >> 1;                  // repeat(,2) along last dim
    float pos, scale;
    if (kk < 192) { pos = -1.f + 2.f * i / 31.f; scale = 1.f + 4.f * kk / 191.f; }
    else          { pos = -1.f + 2.f * j / 31.f; scale = 1.f + 4.f * (kk - 192) / 191.f; }
    S[(size_t)t * DIMD + c] = to_bf16(sinf(pos * scale * 3.14159265358979323846f));
  }
}

// ---------- patchify: x (32,3,512,512) -> P (32768 x 768) bf16, x-coalesced ----------
__global__ __launch_bounds__(256) void k_patches(const float* __restrict__ x,
                                                 __bf16* __restrict__ P) {
  size_t idx = (size_t)blockIdx.x * 256 + threadIdx.x;   // 32*3*512*512 total
  int w = (int)(idx & 511);
  int h = (int)((idx >> 9) & 511);
  int c = (int)((idx >> 18) % 3);
  int b = (int)((idx >> 18) / 3);
  int t = (h >> 4) * 32 + (w >> 4);
  int k = ((h & 15) * 16 + (w & 15)) * 3 + c;
  P[((size_t)(b * NTOK + t)) * DIMD + k] = to_bf16(x[idx]);
}

// ---------- generic WMMA GEMM: C(MxN,f32) = A(MxK,bf16) @ Bt(NxK,bf16)^T ----------
// WG = 8 waves, 4(M)x2(N); wave tile 32x64 -> WG tile 128x128.
enum { EPI_BIAS = 0, EPI_RESID = 1 };

template<int EPI>
__global__ __launch_bounds__(256) void k_gemm(const __bf16* __restrict__ A,
                                              const __bf16* __restrict__ Bt,
                                              const float* __restrict__ bias,
                                              float* __restrict__ C,
                                              int M, int N, int K) {
  const int lane = threadIdx.x & 31;
  const int wave = threadIdx.x >> 5;
  const int bm = blockIdx.y * 128 + (wave >> 1) * 32;
  const int bn = blockIdx.x * 128 + (wave & 1) * 64;
  v8f acc[2][4] = {};
  for (int k = 0; k < K; k += 32) {
    v16bf a0 = load_frag(A, K, bm,      k, lane);
    v16bf a1 = load_frag(A, K, bm + 16, k, lane);
    v16bf b0 = load_frag(Bt, K, bn,      k, lane);
    v16bf b1 = load_frag(Bt, K, bn + 16, k, lane);
    v16bf b2 = load_frag(Bt, K, bn + 32, k, lane);
    v16bf b3 = load_frag(Bt, K, bn + 48, k, lane);
    acc[0][0] = __builtin_amdgcn_wmma_f32_16x16x32_bf16(false, a0, false, b0, (short)0, acc[0][0], false, false);
    acc[0][1] = __builtin_amdgcn_wmma_f32_16x16x32_bf16(false, a0, false, b1, (short)0, acc[0][1], false, false);
    acc[0][2] = __builtin_amdgcn_wmma_f32_16x16x32_bf16(false, a0, false, b2, (short)0, acc[0][2], false, false);
    acc[0][3] = __builtin_amdgcn_wmma_f32_16x16x32_bf16(false, a0, false, b3, (short)0, acc[0][3], false, false);
    acc[1][0] = __builtin_amdgcn_wmma_f32_16x16x32_bf16(false, a1, false, b0, (short)0, acc[1][0], false, false);
    acc[1][1] = __builtin_amdgcn_wmma_f32_16x16x32_bf16(false, a1, false, b1, (short)0, acc[1][1], false, false);
    acc[1][2] = __builtin_amdgcn_wmma_f32_16x16x32_bf16(false, a1, false, b2, (short)0, acc[1][2], false, false);
    acc[1][3] = __builtin_amdgcn_wmma_f32_16x16x32_bf16(false, a1, false, b3, (short)0, acc[1][3], false, false);
  }
  // C layout: lane n = lane&15 ; VGPR v -> row v (lanes 0-15) / v+8 (lanes 16-31)
  #pragma unroll
  for (int i = 0; i < 2; ++i)
    #pragma unroll
    for (int j = 0; j < 4; ++j) {
      int n  = bn + j * 16 + (lane & 15);
      int mb = bm + i * 16 + ((lane >> 4) << 3);
      #pragma unroll
      for (int v = 0; v < 8; ++v) {
        size_t idx = (size_t)(mb + v) * N + n;
        float val = acc[i][j][v];
        if (EPI == EPI_BIAS) val += bias[n];
        else                 val += C[idx];
        C[idx] = val;
      }
    }
}

// ---------- fused swiglu GEMM: H = silu(Z@w1) * (Z@w3), bf16 out ----------
// WG = 8 waves, 4(M)x2(N); wave tile 32x32 dual-accumulated -> WG tile 128x64.
__global__ __launch_bounds__(256) void k_gemm_swiglu(const __bf16* __restrict__ A,
                                                     const __bf16* __restrict__ B1t,
                                                     const __bf16* __restrict__ B3t,
                                                     __bf16* __restrict__ H,
                                                     int M, int N, int K) {
  const int lane = threadIdx.x & 31;
  const int wave = threadIdx.x >> 5;
  const int bm = blockIdx.y * 128 + (wave >> 1) * 32;
  const int bn = blockIdx.x * 64 + (wave & 1) * 32;
  v8f acc1[2][2] = {}, acc3[2][2] = {};
  for (int k = 0; k < K; k += 32) {
    v16bf a0  = load_frag(A,   K, bm,      k, lane);
    v16bf a1  = load_frag(A,   K, bm + 16, k, lane);
    v16bf p0  = load_frag(B1t, K, bn,      k, lane);
    v16bf p1  = load_frag(B1t, K, bn + 16, k, lane);
    v16bf q0  = load_frag(B3t, K, bn,      k, lane);
    v16bf q1  = load_frag(B3t, K, bn + 16, k, lane);
    acc1[0][0] = __builtin_amdgcn_wmma_f32_16x16x32_bf16(false, a0, false, p0, (short)0, acc1[0][0], false, false);
    acc1[0][1] = __builtin_amdgcn_wmma_f32_16x16x32_bf16(false, a0, false, p1, (short)0, acc1[0][1], false, false);
    acc1[1][0] = __builtin_amdgcn_wmma_f32_16x16x32_bf16(false, a1, false, p0, (short)0, acc1[1][0], false, false);
    acc1[1][1] = __builtin_amdgcn_wmma_f32_16x16x32_bf16(false, a1, false, p1, (short)0, acc1[1][1], false, false);
    acc3[0][0] = __builtin_amdgcn_wmma_f32_16x16x32_bf16(false, a0, false, q0, (short)0, acc3[0][0], false, false);
    acc3[0][1] = __builtin_amdgcn_wmma_f32_16x16x32_bf16(false, a0, false, q1, (short)0, acc3[0][1], false, false);
    acc3[1][0] = __builtin_amdgcn_wmma_f32_16x16x32_bf16(false, a1, false, q0, (short)0, acc3[1][0], false, false);
    acc3[1][1] = __builtin_amdgcn_wmma_f32_16x16x32_bf16(false, a1, false, q1, (short)0, acc3[1][1], false, false);
  }
  #pragma unroll
  for (int i = 0; i < 2; ++i)
    #pragma unroll
    for (int j = 0; j < 2; ++j) {
      int n  = bn + j * 16 + (lane & 15);
      int mb = bm + i * 16 + ((lane >> 4) << 3);
      #pragma unroll
      for (int v = 0; v < 8; ++v) {
        float x1 = acc1[i][j][v];
        float x3 = acc3[i][j][v];
        float s  = x1 / (1.f + expf(-x1));          // silu
        H[(size_t)(mb + v) * N + n] = to_bf16(s * x3);
      }
    }
}

// ---------- batch-1 MQA: one block per (query row, head) ----------
__global__ __launch_bounds__(256) void k_mqa(const float* __restrict__ qkv,
                                             __bf16* __restrict__ outb) {
  int t = blockIdx.x, h = blockIdx.y;
  __shared__ float q[HD];
  __shared__ float red[256];
  __shared__ float probs[NTOK];
  int tid = threadIdx.x;
  if (tid < HD) q[tid] = qkv[(size_t)t * QKVN + h * HD + tid];
  __syncthreads();

  float lmax = -3.0e38f;
  for (int j = tid; j <= t; j += 256) {
    const float* kj = qkv + (size_t)j * QKVN + DIMD;
    float s = 0.f;
    #pragma unroll 8
    for (int d = 0; d < HD; ++d) s += q[d] * kj[d];
    s *= 0.125f;                                    // 1/sqrt(64)
    probs[j] = s;
    lmax = fmaxf(lmax, s);
  }
  red[tid] = lmax; __syncthreads();
  for (int off = 128; off > 0; off >>= 1) {
    if (tid < off) red[tid] = fmaxf(red[tid], red[tid + off]);
    __syncthreads();
  }
  float mval = red[0]; __syncthreads();

  float lsum = 0.f;
  for (int j = tid; j <= t; j += 256) { float e = expf(probs[j] - mval); probs[j] = e; lsum += e; }
  red[tid] = lsum; __syncthreads();
  for (int off = 128; off > 0; off >>= 1) {
    if (tid < off) red[tid] += red[tid + off];
    __syncthreads();
  }
  float inv = 1.f / red[0]; __syncthreads();

  int d = tid & 63, g = tid >> 6;                   // 4-way split over j
  float o = 0.f;
  for (int j = g; j <= t; j += 4)
    o += probs[j] * qkv[(size_t)j * QKVN + DIMD + HD + d];
  red[tid] = o; __syncthreads();
  if (g == 0) {
    float r = red[d] + red[64 + d] + red[128 + d] + red[192 + d];
    outb[(size_t)t * DIMD + h * HD + d] = to_bf16(r * inv);
  }
}

// ---------- z_in = tok + att2 (broadcast) -> d_out ; LN(z_in) -> zln bf16 ----------
__global__ __launch_bounds__(256) void k_add_ln(float* __restrict__ zio,   // d_out: in tok, out z_in
                                                const float* __restrict__ att2,
                                                const float* __restrict__ g,
                                                const float* __restrict__ b,
                                                __bf16* __restrict__ zln) {
  int m = blockIdx.x;                 // 0..32767
  int t = m & (NTOK - 1);
  __shared__ float red[256];
  int tid = threadIdx.x;
  float vals[3];
  float s = 0.f;
  #pragma unroll
  for (int ii = 0; ii < 3; ++ii) {
    int c = tid + ii * 256;
    float v = zio[(size_t)m * DIMD + c] + att2[(size_t)t * DIMD + c];
    vals[ii] = v;
    zio[(size_t)m * DIMD + c] = v;
    s += v;
  }
  red[tid] = s; __syncthreads();
  for (int off = 128; off > 0; off >>= 1) { if (tid < off) red[tid] += red[tid + off]; __syncthreads(); }
  float mu = red[0] * (1.f / DIMD); __syncthreads();
  float s2 = 0.f;
  #pragma unroll
  for (int ii = 0; ii < 3; ++ii) { float dd = vals[ii] - mu; s2 += dd * dd; }
  red[tid] = s2; __syncthreads();
  for (int off = 128; off > 0; off >>= 1) { if (tid < off) red[tid] += red[tid + off]; __syncthreads(); }
  float invs = rsqrtf(red[0] * (1.f / DIMD) + 1e-5f);
  #pragma unroll
  for (int ii = 0; ii < 3; ++ii) {
    int c = tid + ii * 256;
    zln[(size_t)m * DIMD + c] = to_bf16((vals[ii] - mu) * invs * g[c] + b[c]);
  }
}

// ---------- launcher ----------
extern "C" void kernel_launch(void* const* d_in, const int* in_sizes, int n_in,
                              void* d_out, int out_size, void* d_ws, size_t ws_size,
                              hipStream_t stream) {
  const float* x      = (const float*)d_in[0];
  const float* patchw = (const float*)d_in[1];
  const float* patchb = (const float*)d_in[2];
  const float* lng    = (const float*)d_in[3];
  const float* lnb    = (const float*)d_in[4];
  const float* w1     = (const float*)d_in[5];
  const float* w2     = (const float*)d_in[6];
  const float* w3     = (const float*)d_in[7];
  const float* wqkv   = (const float*)d_in[8];
  const float* bqkv   = (const float*)d_in[9];
  const float* wo     = (const float*)d_in[10];
  const float* bo     = (const float*)d_in[11];
  float* out = (float*)d_out;

  char* ws = (char*)d_ws;
  size_t off = 0;
  auto alloc = [&](size_t bytes) -> void* {
    void* p = ws + off;
    off = (off + bytes + 255) & ~(size_t)255;
    return p;
  };

  __bf16* pw_t   = (__bf16*)alloc((size_t)DIMD * DIMD * 2);   // 768 x 768 (NxK)
  __bf16* wqkv_t = (__bf16*)alloc((size_t)QKVN * DIMD * 2);   // 896 x 768
  __bf16* wo_t   = (__bf16*)alloc((size_t)DIMD * DIMD * 2);   // 768 x 768
  __bf16* w1_t   = (__bf16*)alloc((size_t)MLP  * DIMD * 2);   // 3072 x 768
  __bf16* w3_t   = (__bf16*)alloc((size_t)MLP  * DIMD * 2);   // 3072 x 768
  __bf16* w2_t   = (__bf16*)alloc((size_t)DIMD * MLP  * 2);   // 768 x 3072
  __bf16* sinb   = (__bf16*)alloc((size_t)NTOK * DIMD * 2);
  float*  qkvf   = (float*) alloc((size_t)NTOK * QKVN * 4);
  __bf16* mqab   = (__bf16*)alloc((size_t)NTOK * DIMD * 2);
  float*  att2   = (float*) alloc((size_t)NTOK * DIMD * 4);
  __bf16* zln    = (__bf16*)alloc((size_t)MROWS * DIMD * 2);
  __bf16* Hbuf   = (__bf16*)alloc((size_t)MROWS * MLP  * 2);
  __bf16* P      = Hbuf;   // alias: patches consumed before Hbuf is written

  dim3 tb(32, 8);
  k_transpose_pack<<<dim3(DIMD/32, DIMD/32), tb, 0, stream>>>(patchw, pw_t,   DIMD, DIMD);
  k_transpose_pack<<<dim3(QKVN/32, DIMD/32), tb, 0, stream>>>(wqkv,   wqkv_t, DIMD, QKVN);
  k_transpose_pack<<<dim3(DIMD/32, DIMD/32), tb, 0, stream>>>(wo,     wo_t,   DIMD, DIMD);
  k_transpose_pack<<<dim3(MLP/32,  DIMD/32), tb, 0, stream>>>(w1,     w1_t,   DIMD, MLP);
  k_transpose_pack<<<dim3(MLP/32,  DIMD/32), tb, 0, stream>>>(w3,     w3_t,   DIMD, MLP);
  k_transpose_pack<<<dim3(DIMD/32, MLP/32),  tb, 0, stream>>>(w2,     w2_t,   MLP,  DIMD);

  k_sin<<<NTOK, 256, 0, stream>>>(sinb);
  k_patches<<<(BATCH * 3 * 512 * 512) / 256, 256, 0, stream>>>(x, P);

  // tok -> d_out (re-used as z_in accumulator)
  k_gemm<EPI_BIAS><<<dim3(DIMD/128, MROWS/128), 256, 0, stream>>>(P, pw_t, patchb, out, MROWS, DIMD, DIMD);
  // qkv = sin @ wqkv + bqkv
  k_gemm<EPI_BIAS><<<dim3(QKVN/128, NTOK/128), 256, 0, stream>>>(sinb, wqkv_t, bqkv, qkvf, NTOK, QKVN, DIMD);
  // multi-query attention (batch 1)
  k_mqa<<<dim3(NTOK, HEADS), 256, 0, stream>>>(qkvf, mqab);
  // att2 = mqa_out @ wo + bo
  k_gemm<EPI_BIAS><<<dim3(DIMD/128, NTOK/128), 256, 0, stream>>>(mqab, wo_t, bo, att2, NTOK, DIMD, DIMD);
  // z_in = tok + att2 -> d_out ; zln = LN(z_in)
  k_add_ln<<<MROWS, 256, 0, stream>>>(out, att2, lng, lnb, zln);
  // H = silu(zln @ w1) * (zln @ w3)
  k_gemm_swiglu<<<dim3(MLP/64, MROWS/128), 256, 0, stream>>>(zln, w1_t, w3_t, Hbuf, MROWS, MLP, DIMD);
  // out = H @ w2 + z_in (residual read-modify-write of d_out)
  k_gemm<EPI_RESID><<<dim3(DIMD/128, MROWS/128), 256, 0, stream>>>(Hbuf, w2_t, nullptr, out, MROWS, DIMD, MLP);
}
